// TernaryConv2d_4320737100213
// MI455X (gfx1250) — matrix operand order repaired and checked by reading, not compiled, not run on hardware
//
#include <hip/hip_runtime.h>
#include <hip/hip_bf16.h>

// ---------------------------------------------------------------------------
// TernaryConv2d on MI455X (gfx1250): implicit GEMM, V_WMMA_F32_16X16X32_BF16.
//   x (32,256,56,56) f32 | weight (256,256,3,3) f32 -> ternary | bias (256,)
//
// GEMM: M=256 (Cout), K=2304 ordered (kh*3+kw)*256+c, N = 1792 rows x 64 pad.
// Per 32-channel slice: stage halo'd x tile ONCE in LDS (channel-pair
// interleaved -> B-fragment dword = one aligned ds_load_b32) and run all
// 9 taps from LDS.  Wave tile 64x64 -> 16 WMMAs per tap, 4 DS loads/WMMA.
// Block = 8 waves tiling N: 64M x 512N (8 padded rows; 8 | 56 so no image
// crossing).  Grid (224, 4).
// ---------------------------------------------------------------------------

typedef __attribute__((ext_vector_type(16))) __bf16 v16bf;
typedef __attribute__((ext_vector_type(8)))  float  v8f;

#define CIN        256
#define COUT       256
#define HW_IMG     3136
#define KDIM       2304
#define NROWS      1792          // 32 images * 56 output rows
#define X_ELEMS    25690112      // 32*256*56*56

// LDS geometry
#define XS_ROWSTR  68            // dwords per (cp,row): slots for w_in=-1..66
#define XS_NROW    10            // 8 output rows + 2 halo
#define XS_CPSTR   684           // 10*68 + 4 pad  (8*684 % 64 == 32 -> +32 banks)
#define AS_COLS    296           // 288 k + 8 pad shorts (row = 592 B, 148 dw)

__device__ __forceinline__ unsigned short f2bf(float f) {
    union { float f; unsigned u; } cv; cv.f = f;
    unsigned r = cv.u + 0x7FFFu + ((cv.u >> 16) & 1u);
    return (unsigned short)(r >> 16);
}

// ---------------------------------------------------------------------------
// Kernel 1: per-filter ternarization -> bf16 A matrix [o][(kh*3+kw)*256 + c]
// ---------------------------------------------------------------------------
__global__ __launch_bounds__(256) void ternarize_kernel(
        const float* __restrict__ w, unsigned short* __restrict__ wt) {
    const int o   = blockIdx.x;
    const int tid = threadIdx.x;
    __shared__ float red[256];
    __shared__ float red2[256];
    __shared__ float s_delta, s_alpha;
    const float* wf = w + o * KDIM;

    float s = 0.f;
    for (int i = tid; i < KDIM; i += 256) s += fabsf(wf[i]);
    red[tid] = s; __syncthreads();
    for (int off = 128; off > 0; off >>= 1) {
        if (tid < off) red[tid] += red[tid + off];
        __syncthreads();
    }
    if (tid == 0) s_delta = (0.7f / (float)KDIM) * red[0];
    __syncthreads();
    const float delta = s_delta;

    float s2 = 0.f, c2 = 0.f;
    for (int i = tid; i < KDIM; i += 256) {
        float a = fabsf(wf[i]);
        if (a > delta) { s2 += a; c2 += 1.f; }
    }
    red[tid] = s2; red2[tid] = c2; __syncthreads();
    for (int off = 128; off > 0; off >>= 1) {
        if (tid < off) { red[tid] += red[tid + off]; red2[tid] += red2[tid + off]; }
        __syncthreads();
    }
    if (tid == 0) s_alpha = red[0] / red2[0];
    __syncthreads();
    const float alpha = s_alpha;

    for (int i = tid; i < KDIM; i += 256) {
        float v = wf[i];
        float t = (v > delta) ? alpha : ((v < -delta) ? -alpha : 0.f);
        int c = i / 9, r = i - c * 9;
        wt[o * KDIM + r * 256 + c] = f2bf(t);
    }
}

// ---------------------------------------------------------------------------
// Kernel 2: x f32 -> bf16 (same NCHW layout)
// ---------------------------------------------------------------------------
__global__ __launch_bounds__(256) void convert_x_kernel(
        const float* __restrict__ x, unsigned short* __restrict__ xb, int n) {
    int i = (blockIdx.x * 256 + threadIdx.x) * 4;
    if (i + 3 < n) {
        float4 v = *(const float4*)(x + i);
        union { unsigned short s[4]; uint2 q; } p;
        p.s[0] = f2bf(v.x); p.s[1] = f2bf(v.y);
        p.s[2] = f2bf(v.z); p.s[3] = f2bf(v.w);
        *(uint2*)(xb + i) = p.q;
    }
}

// ---------------------------------------------------------------------------
// stage one half-row of the halo'd x tile (2 channels interleaved per dword)
// ---------------------------------------------------------------------------
__device__ __forceinline__ void stage_xrow(
        const unsigned short* __restrict__ xb, unsigned int* __restrict__ Xs,
        int img, int c0, int cp, int ridx, int halfw, int h0) {
    const int  h_in = h0 - 1 + ridx;
    const bool rv   = (unsigned)h_in < 56u;
    const unsigned short* se =
        xb + (((size_t)img * CIN + c0 + 2 * cp) * 56 + (rv ? h_in : 0)) * 56;
    const int xwb  = halfw * 34;
    const int base = cp * XS_CPSTR + ridx * XS_ROWSTR + xwb;
    #pragma unroll
    for (int s = 0; s < 34; ++s) {
        const int  w_in = xwb + s - 1;                 // slot = w_in + 1
        const bool v    = rv && ((unsigned)w_in < 56u);
        unsigned lo = v ? (unsigned)se[w_in]          : 0u;
        unsigned hi = v ? (unsigned)se[w_in + HW_IMG] : 0u;
        Xs[base + s] = lo | (hi << 16);
    }
}

// ---------------------------------------------------------------------------
// Kernel 3: conv.  grid = (224, 4), block = 256 (8 waves tiling N).
//   Block tile: M=64 channels, N=512 = 8 padded rows of 64 cols (real 56).
//   Wave tile 64x64: 4 A frags x 4 B frags -> 16 WMMAs per tap.
// ---------------------------------------------------------------------------
__global__ __launch_bounds__(256) void conv_wmma_kernel(
        const unsigned short* __restrict__ xb,
        const unsigned short* __restrict__ wt,
        const float* __restrict__ bias,
        float* __restrict__ out) {
    __shared__ unsigned short As[64][AS_COLS];       // 37,888 B
    __shared__ unsigned int   Xs[16 * XS_CPSTR];     // 43,776 B {c_even,c_odd}

    const int tid  = threadIdx.x;
    const int lane = tid & 31;
    const int wv   = tid >> 5;        // 0..7 : wave's 64-col N strip
    const int lh   = lane >> 4;       // WMMA lane-half
    const int lr   = lane & 15;

    const int M0  = blockIdx.y * 64;
    const int RB  = blockIdx.x * 8;   // 8 output rows, always same image
    const int img = RB / 56;
    const int h0  = RB - img * 56;

    // X staging mapping: cp = tid&15; unit = tid>>4 covers (ridx 0..7, half)
    const int xcp   = tid & 15;
    const int xunit = tid >> 4;            // 0..15
    const int xridx = xunit >> 1;          // 0..7
    const int xhalf = xunit & 1;

    // A staging mapping: 64 m x 4 chunks of 8 k
    const int am  = tid >> 2;
    const int ak8 = (tid & 3) << 3;

    v8f acc[4][4] = {};

    for (int cs = 0; cs < 8; ++cs) {
        const int c0 = cs * 32;
        __syncthreads();   // previous slice's LDS reads complete

        // ---- stage A: 9 taps x 32 ch for 64 output channels ----
        {
            const unsigned short* arow = wt + (size_t)(M0 + am) * KDIM + c0 + ak8;
            #pragma unroll
            for (int r = 0; r < 9; ++r)
                *(uint4*)&As[am][r * 32 + ak8] = *(const uint4*)(arow + r * 256);
        }

        // ---- stage X: 10 halo rows, channel-pair interleaved ----
        stage_xrow(xb, Xs, img, c0, xcp, xridx, xhalf, h0);
        if (xunit < 4)   // units 0..3 pick up halo rows 8,9
            stage_xrow(xb, Xs, img, c0, xcp, 8 + (xunit >> 1), xunit & 1, h0);

        // prefetch next slice toward L2 while this slice computes
        if (cs < 7) {
            __builtin_prefetch(xb + (((size_t)img * CIN + c0 + 32 + 2 * xcp) * 56 + h0) * 56, 0, 1);
            __builtin_prefetch(wt + (size_t)(M0 + am) * KDIM + c0 + 32, 0, 1);
        }

        __syncthreads();

        // ---- 9 taps from LDS: 16 WMMAs each ----
        #pragma unroll 3
        for (int r = 0; r < 9; ++r) {
            const int kh = r / 3, kw = r - kh * 3;

            union { v16bf v; unsigned u[8]; } afrag[4], bfrag[4];
            #pragma unroll
            for (int i = 0; i < 4; ++i) {
                const int m = i * 16 + lr;
                #pragma unroll
                for (int q = 0; q < 8; ++q) {
                    const int kk = ((q < 4) ? 2 * q : 2 * q + 8) + 8 * lh;
                    afrag[i].u[q] = *(const unsigned*)&As[m][r * 32 + kk];
                }
            }
            #pragma unroll
            for (int j = 0; j < 4; ++j) {
                const int n    = wv * 64 + j * 16 + lr;
                const int ridx = (n >> 6) + kh;              // input row in tile
                const int slot = (n & 63) + kw;              // w_in + 1
                #pragma unroll
                for (int q = 0; q < 8; ++q) {
                    const int cp = ((q < 4) ? q : q + 4) + 4 * lh;
                    bfrag[j].u[q] = Xs[cp * XS_CPSTR + ridx * XS_ROWSTR + slot];
                }
            }
            #pragma unroll
            for (int i = 0; i < 4; ++i)
                #pragma unroll
                for (int j = 0; j < 4; ++j)
                    acc[i][j] = __builtin_amdgcn_wmma_f32_16x16x32_bf16(
                        false, afrag[i].v, false, bfrag[j].v,
                        (short)0, acc[i][j], false, false);
        }
    }

    // ---- epilogue: bias add, mask padded cols, NCHW stores ----
    #pragma unroll
    for (int i = 0; i < 4; ++i) {
        #pragma unroll
        for (int j = 0; j < 4; ++j) {
            const int n    = wv * 64 + j * 16 + lr;
            const int rsel = n >> 6;
            const int col  = n & 63;
            if (col >= 56) continue;
            const int hh = h0 + rsel;
            #pragma unroll
            for (int q = 0; q < 8; ++q) {
                const int o = M0 + i * 16 + 8 * lh + q;
                size_t idx = (((size_t)img * COUT + o) * 56 + hh) * 56 + col;
                out[idx] = acc[i][j][q] + bias[o];
            }
        }
    }
}

// ---------------------------------------------------------------------------
extern "C" void kernel_launch(void* const* d_in, const int* in_sizes, int n_in,
                              void* d_out, int out_size, void* d_ws, size_t ws_size,
                              hipStream_t stream) {
    const float* x    = (const float*)d_in[0];
    const float* w    = (const float*)d_in[1];
    const float* bias = (const float*)d_in[2];
    float*       out  = (float*)d_out;

    unsigned short* xbuf = (unsigned short*)d_ws;
    unsigned short* wtr  = xbuf + X_ELEMS;       // +51,380,224 B

    ternarize_kernel<<<COUT, 256, 0, stream>>>(w, wtr);
    convert_x_kernel<<<X_ELEMS / 1024, 256, 0, stream>>>(x, xbuf, X_ELEMS);

    dim3 grid(NROWS / 8, COUT / 64);             // (224, 4)
    conv_wmma_kernel<<<grid, 256, 0, stream>>>(xbuf, wtr, bias, out);
}
